// ResidualIntegrationNetworkRK4_15710990369029
// MI455X (gfx1250) — matrix operand order repaired and checked by reading, not compiled, not run on hardware
//
#include <hip/hip_runtime.h>

#define SEQ   8192
#define BATCH 4096
#define HID   128
#define DTC   1.0f

typedef __attribute__((ext_vector_type(16))) _Float16 v16h;
typedef __attribute__((ext_vector_type(8)))  _Float16 v8h;
typedef __attribute__((ext_vector_type(8)))  float    v8f;

__device__ __forceinline__ float tanh_fast(float x) {
#if __has_builtin(__builtin_amdgcn_tanhf)
  return __builtin_amdgcn_tanhf(x);          // v_tanh_f32 (gfx1250 TRANS unit)
#else
  float e = __expf(2.0f * x);                // v_exp_f32
  return 1.0f - 2.0f * __builtin_amdgcn_rcpf(e + 1.0f);
#endif
}

// One block (4 waves / 128 threads) owns a 16-row batch tile for the entire
// 8192-step sequential RK4 rollout. Wave w owns output columns [32w, 32w+32)
// of the 128x128 hidden GEMM (2 WMMA N-tiles), with W2 B-fragments resident
// in VGPRs. Grid = 4096/16 = 256 blocks.
__global__ __launch_bounds__(128) void rk4_ode_kernel(
    const float* __restrict__ x,  const float* __restrict__ W1,
    const float* __restrict__ b1, const float* __restrict__ W2,
    const float* __restrict__ b2, const float* __restrict__ W3,
    const float* __restrict__ b3, float* __restrict__ out)
{
  // h1 tile: 16 rows x 128 f16, padded to 136 f16/row (272B) -> conflict-free
  __shared__ alignas(16) _Float16 h1s[16 * 136];
  // layer-3 partial dot reduction: 16 rows x 64 slots, padded to 68 dwords
  __shared__ alignas(16) float    kred[16 * 68];

  const int lane = threadIdx.x & 31;
  const int wave = threadIdx.x >> 5;
  const int m    = lane & 15;       // batch row within tile (A fragment M, C fragment N)
  const int hlf  = lane >> 4;       // lane half (K/M split per WMMA layouts)
  const int rowbase = blockIdx.x * 16;

  // ---- W1/b1 slice: lane computes h1[m][jb .. jb+15] each stage ----
  const int jb = wave * 32 + hlf * 16;
  float w1a[16], w1b[16], b1r[16];
#pragma unroll
  for (int q = 0; q < 16; ++q) {
    w1a[q] = W1[jb + q];           // W1[0][j] : input weight
    w1b[q] = W1[HID + jb + q];     // W1[1][j] : state weight
    b1r[q] = b1[jb + q];
  }

  // ---- W2 -> resident B fragments (f16), 2 N-tiles x 4 K-frags ----
  // B 32x16 f16, this code's contiguous K order: lane = N (l&15); lanes 0-15 hold K 0..15, lanes
  // 16-31 hold K 16..31; element e of v16h = K (16*hlf + e).
  v16h Bf[2][4];
  float b2r[2], w3r[2];
#pragma unroll
  for (int nt = 0; nt < 2; ++nt) {
    const int n = (2 * wave + nt) * 16 + m;   // global output column
#pragma unroll
    for (int kf = 0; kf < 4; ++kf) {
      const int kb = kf * 32 + hlf * 16;
      v16h bf;
#pragma unroll
      for (int e = 0; e < 16; ++e)
        bf[e] = (_Float16)W2[(kb + e) * HID + n];
      Bf[nt][kf] = bf;
    }
    b2r[nt] = b2[n];
    w3r[nt] = W3[n];
  }
  const float b3v = b3[0];

  float y = 0.0f;                   // state y[m], replicated in both halves

  for (int t = 0; t < SEQ; ++t) {
    const float v = x[t * BATCH + rowbase + m];
    if (t + 1 < SEQ)
      __builtin_prefetch(&x[(t + 1) * BATCH + rowbase + m], 0, 1);

    float ys   = y;                 // stage input state
    float ksum = 0.0f;              // dt/6 * (k1 + 2k2 + 2k3 + k4)

#pragma unroll
    for (int s = 0; s < 4; ++s) {
      // ---------------- layer 1: h1 = tanh(v*W1a + ys*W1b + b1) ----------
      v8h p0, p1;
#pragma unroll
      for (int q = 0; q < 8; ++q) {
        p0[q] = (_Float16)tanh_fast(fmaf(v, w1a[q],     fmaf(ys, w1b[q],     b1r[q])));
        p1[q] = (_Float16)tanh_fast(fmaf(v, w1a[q + 8], fmaf(ys, w1b[q + 8], b1r[q + 8])));
      }
      *(v8h*)(h1s + m * 136 + jb)     = p0;   // ds_store_b128
      *(v8h*)(h1s + m * 136 + jb + 8) = p1;
      __syncthreads();

      // ---------------- build A fragments (contiguous K order, as B) --------
      // elements 0..7  : K = 32*kf + 16*hlf + 0..7
      // elements 8..15 : K = 32*kf + 16*hlf + 8 + 0..7
      v16h A[4];
#pragma unroll
      for (int kf = 0; kf < 4; ++kf) {
        v8h lo = *(const v8h*)(h1s + m * 136 + kf * 32 + hlf * 16);
        v8h hi = *(const v8h*)(h1s + m * 136 + kf * 32 + hlf * 16 + 8);
        A[kf] = __builtin_shufflevector(lo, hi,
                  0, 1, 2, 3, 4, 5, 6, 7, 8, 9, 10, 11, 12, 13, 14, 15);
      }

      // ---------------- layer 2 (WMMA) fused with layer-3 dot ------------
      float acc[8];
#pragma unroll
      for (int r = 0; r < 8; ++r) acc[r] = 0.0f;
#pragma unroll
      for (int nt = 0; nt < 2; ++nt) {
        v8f C = {0.f, 0.f, 0.f, 0.f, 0.f, 0.f, 0.f, 0.f};
#pragma unroll
        for (int kf = 0; kf < 4; ++kf)
          C = __builtin_amdgcn_wmma_f32_16x16x32_f16(
                false, A[kf], false, Bf[nt][kf], (short)0, C, false, false);
        // C element r: M = r + 8*hlf, N = 16*(2*wave+nt) + m
#pragma unroll
        for (int r = 0; r < 8; ++r)
          acc[r] = fmaf(tanh_fast(C[r] + b2r[nt]), w3r[nt], acc[r]);
      }

      // ---------------- reduce layer-3 partials across lanes/waves -------
#pragma unroll
      for (int r = 0; r < 8; ++r) {
        const int mr = r + 8 * hlf;                 // row this partial belongs to
        kred[mr * 68 + wave * 16 + m] = acc[r];
      }
      __syncthreads();
      float ks = b3v;
      const float4* kp = (const float4*)(kred + m * 68);
#pragma unroll
      for (int i = 0; i < 16; ++i) {
        float4 f4 = kp[i];
        ks += (f4.x + f4.y) + (f4.z + f4.w);
      }

      // ---------------- RK4 combine --------------------------------------
      const float cw  = (s == 0 || s == 3) ? (DTC / 6.0f) : (DTC / 3.0f);
      const float cin = (s == 0) ? 0.5f * DTC
                      : (s == 1) ? 0.5f * DTC
                      : (s == 2) ? DTC : 0.0f;
      ksum = fmaf(cw, ks, ksum);
      ys   = fmaf(cin, ks, y);      // input state for next stage
    }

    y += ksum;
    if (threadIdx.x < 16)
      out[t * BATCH + rowbase + threadIdx.x] = y;  // one wave writes the tile
  }
}

extern "C" void kernel_launch(void* const* d_in, const int* in_sizes, int n_in,
                              void* d_out, int out_size, void* d_ws, size_t ws_size,
                              hipStream_t stream) {
  (void)in_sizes; (void)n_in; (void)out_size; (void)d_ws; (void)ws_size;
  const float* x  = (const float*)d_in[0];  // (SEQ, BATCH, 1)
  const float* W1 = (const float*)d_in[1];  // (2, HID)
  const float* b1 = (const float*)d_in[2];  // (HID,)
  const float* W2 = (const float*)d_in[3];  // (HID, HID)
  const float* b2 = (const float*)d_in[4];  // (HID,)
  const float* W3 = (const float*)d_in[5];  // (HID, 1)
  const float* b3 = (const float*)d_in[6];  // (1,)
  float* out = (float*)d_out;               // (SEQ, BATCH, 1)

  dim3 grid(BATCH / 16);   // 256 row-tiles, persistent over all 8192 steps
  dim3 block(128);         // 4 wave32s per tile
  rk4_ode_kernel<<<grid, block, 0, stream>>>(x, W1, b1, W2, b2, W3, b3, out);
}
